// MixtureOfAttentionHeads_38774964748494
// MI455X (gfx1250) — matrix operand order repaired and checked by reading, not compiled, not run on hardware
//
#include <hip/hip_runtime.h>
#include <hip/hip_bf16.h>

typedef __attribute__((ext_vector_type(16))) __bf16 v16bf;
typedef __attribute__((ext_vector_type(8)))  __bf16 v8bf;
typedef __attribute__((ext_vector_type(2)))  __bf16 v2bf;
typedef __attribute__((ext_vector_type(8)))  float  v8f;

#define T_TOK    8192
#define D_DIM    1024
#define H_DIM    1024
#define DOUT_DIM 1024
#define E_CNT    8
#define K_TOP    2
#define NSLOTS   (T_TOK * K_TOP)          // 16384 (token, k) pairs
#define BM       128                       // rows per block tile
#define BN       64                        // cols per block tile
#define NT       4                         // 16-wide n-subtiles per wave
#define MAXT     ((NSLOTS + BM - 1) / BM)  // 128 row tiles per expert (upper bound)

// ---- float -> bf16 conversion helpers ------------------------------------
// Prefer the hardware packed converter (1 instr / 2 elements); fall back to a
// manual round-to-nearest-even sequence if the builtin is unavailable.
static __device__ __forceinline__ __bf16 f2bf_manual(float f) {
    unsigned u = __float_as_uint(f);
    unsigned r = (u + 0x7FFFu + ((u >> 16) & 1u)) >> 16;
    unsigned short s = (unsigned short)r;
    return __builtin_bit_cast(__bf16, s);
}

static __device__ __forceinline__ v2bf pk_bf16(float a, float b) {
#if __has_builtin(__builtin_amdgcn_cvt_pk_bf16_f32)
    return __builtin_amdgcn_cvt_pk_bf16_f32(a, b);
#else
    v2bf r; r[0] = f2bf_manual(a); r[1] = f2bf_manual(b); return r;
#endif
}

static __device__ __forceinline__ __bf16 f2bf(float f) {
#if __has_builtin(__builtin_amdgcn_cvt_pk_bf16_f32)
    return __builtin_amdgcn_cvt_pk_bf16_f32(f, 0.f)[0];
#else
    return f2bf_manual(f);
#endif
}

// ---------------------------------------------------------------- utilities
__global__ void zero_counts_kernel(int* counts) {
    if (threadIdx.x < E_CNT) counts[threadIdx.x] = 0;
}

// x [T][D] f32 -> xb [T][D] bf16 (pure streaming pass, packed converts)
__global__ void conv_x_kernel(const float* __restrict__ x, __bf16* __restrict__ xb) {
    int i = blockIdx.x * 256 + threadIdx.x;          // T*D/2 pairs
    float2 f = ((const float2*)x)[i];
    ((v2bf*)xb)[i] = pk_bf16(f.x, f.y);
}

// Wi [E][D][H] f32 -> Wib [E][H][D] bf16 (transposed so K is contiguous per column)
__global__ void conv_wi_kernel(const float* __restrict__ Wi, __bf16* __restrict__ Wib) {
    int i = blockIdx.x * 256 + threadIdx.x;          // E*D*H/2 pairs
    int o   = i * 2;                                 // output linear index (d even)
    int e   = o >> 20;
    int rem = o & ((1 << 20) - 1);
    int h   = rem >> 10;
    int d   = rem & 1023;
    size_t src = ((size_t)e << 20) + ((size_t)d << 10) + h;  // Wi[e][d][h]
    ((v2bf*)Wib)[i] = pk_bf16(Wi[src], Wi[src + 1024]);      // d, d+1
}

// Wo [E][H][DOUT] f32 -> Wob [E][DOUT][H] bf16
__global__ void conv_wo_kernel(const float* __restrict__ Wo, __bf16* __restrict__ Wob) {
    int i = blockIdx.x * 256 + threadIdx.x;          // E*DOUT*H/2 pairs
    int o    = i * 2;                                // output linear index (h even)
    int e    = o >> 20;
    int rem  = o & ((1 << 20) - 1);
    int dout = rem >> 10;
    int h    = rem & 1023;
    size_t src = ((size_t)e << 20) + ((size_t)h << 10) + dout; // Wo[e][h][dout]
    ((v2bf*)Wob)[i] = pk_bf16(Wo[src], Wo[src + 1024]);        // h, h+1
}

// ---------------------------------------------------------------- router
// One wave32 per token: logits (8 experts), softmax, top-2, scatter to lists.
__global__ __launch_bounds__(256) void moe_router_kernel(
    const float* __restrict__ x, const float* __restrict__ Wr,
    float* __restrict__ gates, int* __restrict__ lists, int* __restrict__ counts)
{
    int wave = threadIdx.x >> 5;
    int lane = threadIdx.x & 31;
    int t = blockIdx.x * 8 + wave;
    const float* xr = x + (size_t)t * D_DIM;

    float acc[E_CNT];
#pragma unroll
    for (int e = 0; e < E_CNT; e++) acc[e] = 0.f;

    for (int d = lane; d < D_DIM; d += 32) {
        float xv = xr[d];
        const float4* wr = (const float4*)(Wr + (size_t)d * E_CNT);
        float4 w0 = wr[0], w1 = wr[1];
        acc[0] += xv * w0.x; acc[1] += xv * w0.y; acc[2] += xv * w0.z; acc[3] += xv * w0.w;
        acc[4] += xv * w1.x; acc[5] += xv * w1.y; acc[6] += xv * w1.z; acc[7] += xv * w1.w;
    }
#pragma unroll
    for (int e = 0; e < E_CNT; e++)
        for (int off = 16; off >= 1; off >>= 1)
            acc[e] += __shfl_xor(acc[e], off, 32);

    if (lane == 0) {
        float mx = acc[0];
#pragma unroll
        for (int e = 1; e < E_CNT; e++) mx = fmaxf(mx, acc[e]);
        float p[E_CNT], sum = 0.f;
#pragma unroll
        for (int e = 0; e < E_CNT; e++) { p[e] = __expf(acc[e] - mx); sum += p[e]; }
        float inv = 1.f / sum;
        int i0 = 0;
#pragma unroll
        for (int e = 1; e < E_CNT; e++) if (p[e] > p[i0]) i0 = e;
        int i1 = (i0 == 0) ? 1 : 0;
#pragma unroll
        for (int e = 0; e < E_CNT; e++) if (e != i0 && p[e] > p[i1]) i1 = e;

        gates[t * 2 + 0] = p[i0] * inv;   // softmax prob (reference does not renormalize)
        gates[t * 2 + 1] = p[i1] * inv;
        int pos0 = atomicAdd(&counts[i0], 1);
        lists[i0 * NSLOTS + pos0] = t * 2;
        int pos1 = atomicAdd(&counts[i1], 1);
        lists[i1 * NSLOTS + pos1] = t * 2 + 1;
    }
}

// ---------------------------------------------------------------- GEMM1
// h[slot] = xb[token(slot)] @ Wi[e], bf16 WMMA, f32 accumulate, output bf16.
__global__ __launch_bounds__(256) void moe_gemm1_kernel(
    const __bf16* __restrict__ xb, const __bf16* __restrict__ Wib,  // [E][H][D]
    const int* __restrict__ lists, const int* __restrict__ counts,
    __bf16* __restrict__ hb)                                        // [NSLOTS][H]
{
    int e    = blockIdx.x / MAXT;
    int tile = blockIdx.x - e * MAXT;
    int cnt  = counts[e];
    int row0 = tile * BM;
    if (row0 >= cnt) return;
    int n0 = blockIdx.y * BN;

    int lane = threadIdx.x & 31;
    int wave = threadIdx.x >> 5;
    int hi   = lane >> 4;       // lane-half selects K sub-range per ISA layout
    int mlo  = lane & 15;
    const int* mylist = lists + e * NSLOTS;

    // A: lane holds row m = mlo; elems 0-7 -> K = k0+8*hi+j, elems 8-15 -> +16
    int r  = row0 + wave * 16 + mlo;
    int rc = r < cnt ? r : cnt - 1;
    int slot = mylist[rc];
    const __bf16* aptr = xb + (size_t)(slot >> 1) * D_DIM + hi * 8;

    v8f acc[NT];
#pragma unroll
    for (int i = 0; i < NT; i++)
#pragma unroll
        for (int j = 0; j < 8; j++) acc[i][j] = 0.f;

    // B: lane holds col n = mlo; elems 0-7 -> K = k0+8*hi+j, elems 8-15 -> +16
    const __bf16* bptr[NT];
#pragma unroll
    for (int nt = 0; nt < NT; nt++) {
        int col = n0 + nt * 16 + mlo;
        bptr[nt] = Wib + ((size_t)e * H_DIM + col) * D_DIM + hi * 8;
    }

    for (int k0 = 0; k0 < D_DIM; k0 += 32) {
        v8bf a0 = *(const v8bf*)(aptr + k0);
        v8bf a1 = *(const v8bf*)(aptr + k0 + 16);
        v16bf a;
#pragma unroll
        for (int j = 0; j < 8; j++) { a[j] = a0[j]; a[8 + j] = a1[j]; }
#pragma unroll
        for (int nt = 0; nt < NT; nt++) {
            v8bf b0 = *(const v8bf*)(bptr[nt] + k0);
            v8bf b1 = *(const v8bf*)(bptr[nt] + k0 + 16);
            v16bf b;
#pragma unroll
            for (int j = 0; j < 8; j++) { b[j] = b0[j]; b[8 + j] = b1[j]; }
            acc[nt] = __builtin_amdgcn_wmma_f32_16x16x32_bf16(
                false, a, false, b, (short)0, acc[nt], false, false);
        }
    }

    // D layout: lane holds col n = mlo, VGPR v -> row m = v + 8*hi
#pragma unroll
    for (int v = 0; v < 8; v++) {
        int rr = row0 + wave * 16 + v + hi * 8;
        if (rr < cnt) {
            int s2 = mylist[rr];
            __bf16* hrow = hb + (size_t)s2 * H_DIM + n0 + mlo;
#pragma unroll
            for (int nt = 0; nt < NT; nt++) hrow[nt * 16] = f2bf(acc[nt][v]);
        }
    }
}

// ---------------------------------------------------------------- GEMM2
// obuf[slot] = gate[slot] * (h[slot] @ Wo[e])
__global__ __launch_bounds__(256) void moe_gemm2_kernel(
    const __bf16* __restrict__ hb, const __bf16* __restrict__ Wob,  // [E][DOUT][H]
    const int* __restrict__ lists, const int* __restrict__ counts,
    const float* __restrict__ gates, float* __restrict__ obuf)      // [NSLOTS][DOUT]
{
    int e    = blockIdx.x / MAXT;
    int tile = blockIdx.x - e * MAXT;
    int cnt  = counts[e];
    int row0 = tile * BM;
    if (row0 >= cnt) return;
    int n0 = blockIdx.y * BN;

    int lane = threadIdx.x & 31;
    int wave = threadIdx.x >> 5;
    int hi   = lane >> 4;
    int mlo  = lane & 15;
    const int* mylist = lists + e * NSLOTS;

    int r  = row0 + wave * 16 + mlo;
    int rc = r < cnt ? r : cnt - 1;
    int slot = mylist[rc];
    const __bf16* aptr = hb + (size_t)slot * H_DIM + hi * 8;

    v8f acc[NT];
#pragma unroll
    for (int i = 0; i < NT; i++)
#pragma unroll
        for (int j = 0; j < 8; j++) acc[i][j] = 0.f;

    const __bf16* bptr[NT];
#pragma unroll
    for (int nt = 0; nt < NT; nt++) {
        int col = n0 + nt * 16 + mlo;
        bptr[nt] = Wob + ((size_t)e * DOUT_DIM + col) * H_DIM + hi * 8;
    }

    for (int k0 = 0; k0 < H_DIM; k0 += 32) {
        v8bf a0 = *(const v8bf*)(aptr + k0);
        v8bf a1 = *(const v8bf*)(aptr + k0 + 16);
        v16bf a;
#pragma unroll
        for (int j = 0; j < 8; j++) { a[j] = a0[j]; a[8 + j] = a1[j]; }
#pragma unroll
        for (int nt = 0; nt < NT; nt++) {
            v8bf b0 = *(const v8bf*)(bptr[nt] + k0);
            v8bf b1 = *(const v8bf*)(bptr[nt] + k0 + 16);
            v16bf b;
#pragma unroll
            for (int j = 0; j < 8; j++) { b[j] = b0[j]; b[8 + j] = b1[j]; }
            acc[nt] = __builtin_amdgcn_wmma_f32_16x16x32_bf16(
                false, a, false, b, (short)0, acc[nt], false, false);
        }
    }

#pragma unroll
    for (int v = 0; v < 8; v++) {
        int rr = row0 + wave * 16 + v + hi * 8;
        if (rr < cnt) {
            int s2 = mylist[rr];
            float g = gates[s2];
            float* orow = obuf + (size_t)s2 * DOUT_DIM + n0 + mlo;
#pragma unroll
            for (int nt = 0; nt < NT; nt++) orow[nt * 16] = g * acc[nt][v];
        }
    }
}

// ---------------------------------------------------------------- combine
__global__ void moe_combine_kernel(const float* __restrict__ obuf, float* __restrict__ out) {
    int i = blockIdx.x * 256 + threadIdx.x;   // T_TOK * DOUT
    int t = i >> 10;
    int n = i & 1023;
    size_t b = (size_t)(t * 2) * DOUT_DIM + n;
    out[i] = obuf[b] + obuf[b + DOUT_DIM];
}

// ---------------------------------------------------------------- launch
extern "C" void kernel_launch(void* const* d_in, const int* in_sizes, int n_in,
                              void* d_out, int out_size, void* d_ws, size_t ws_size,
                              hipStream_t stream) {
    const float* x  = (const float*)d_in[0];
    const float* Wr = (const float*)d_in[1];
    const float* Wi = (const float*)d_in[2];
    const float* Wo = (const float*)d_in[3];
    float* out = (float*)d_out;

    char* ws = (char*)d_ws;
    size_t off = 0;
    auto alloc = [&](size_t bytes) -> char* {
        char* p = ws + off;
        off += (bytes + 255) & ~(size_t)255;
        return p;
    };
    __bf16* xb    = (__bf16*)alloc((size_t)T_TOK * D_DIM * 2);              // 16 MB
    __bf16* Wib   = (__bf16*)alloc((size_t)E_CNT * D_DIM * H_DIM * 2);      // 16 MB
    __bf16* Wob   = (__bf16*)alloc((size_t)E_CNT * H_DIM * DOUT_DIM * 2);   // 16 MB
    __bf16* hb    = (__bf16*)alloc((size_t)NSLOTS * H_DIM * 2);             // 32 MB
    float*  obuf  = (float*) alloc((size_t)NSLOTS * DOUT_DIM * 4);          // 64 MB
    float*  gates = (float*) alloc((size_t)NSLOTS * 4);
    int*    lists = (int*)   alloc((size_t)E_CNT * NSLOTS * 4);
    int*    counts= (int*)   alloc((size_t)E_CNT * 4);

    hipLaunchKernelGGL(zero_counts_kernel, dim3(1), dim3(32), 0, stream, counts);
    hipLaunchKernelGGL(conv_x_kernel, dim3(T_TOK * D_DIM / 2 / 256), dim3(256), 0, stream, x, xb);
    hipLaunchKernelGGL(conv_wi_kernel, dim3(E_CNT * D_DIM * H_DIM / 2 / 256), dim3(256), 0, stream, Wi, Wib);
    hipLaunchKernelGGL(conv_wo_kernel, dim3(E_CNT * H_DIM * DOUT_DIM / 2 / 256), dim3(256), 0, stream, Wo, Wob);
    hipLaunchKernelGGL(moe_router_kernel, dim3(T_TOK / 8), dim3(256), 0, stream, x, Wr, gates, lists, counts);
    hipLaunchKernelGGL(moe_gemm1_kernel, dim3(E_CNT * MAXT, H_DIM / BN), dim3(256), 0, stream,
                       xb, Wib, lists, counts, hb);
    hipLaunchKernelGGL(moe_gemm2_kernel, dim3(E_CNT * MAXT, DOUT_DIM / BN), dim3(256), 0, stream,
                       hb, Wob, lists, counts, gates, obuf);
    hipLaunchKernelGGL(moe_combine_kernel, dim3(T_TOK * DOUT_DIM / 256), dim3(256), 0, stream, obuf, out);
}
